// MoEAttentionProjection_15204184227977
// MI455X (gfx1250) — matrix operand order repaired and checked by reading, not compiled, not run on hardware
//
#include <hip/hip_runtime.h>
#include <stdint.h>

typedef __attribute__((ext_vector_type(16))) __bf16 v16bf;
typedef __attribute__((ext_vector_type(2)))  __bf16 v2bf;
typedef __attribute__((ext_vector_type(8)))  float  v8f;

#define D_DIM 1024
#define O_DIM 1024
#define E_NUM 8
#define TILE_M 128
#define TILE_N 128
#define TILE_K 32
#define LDS_ROW_DW 20   // 80 bytes/row: 64B data + 16B pad (16B-aligned, conflict-free b128 reads)

__device__ __forceinline__ uint32_t pack_bf16_rne(float a, float b) {
#if __has_builtin(__builtin_amdgcn_cvt_pk_bf16_f32)
    union { v2bf v; uint32_t u; } r;
    r.v = __builtin_amdgcn_cvt_pk_bf16_f32(a, b);
    return r.u;
#else
    uint32_t ua = __float_as_uint(a);
    uint32_t ub = __float_as_uint(b);
    uint32_t ra = (ua + 0x7FFFu + ((ua >> 16) & 1u)) >> 16;
    uint32_t rb = (ub + 0x7FFFu + ((ub >> 16) & 1u)) >> 16;
    return ra | (rb << 16);
#endif
}

// ---------------------------------------------------------------------------
// Kernel 0: zero output + per-expert counters (d_out/d_ws are poisoned 0xAA)
// ---------------------------------------------------------------------------
__global__ void moe_zero_kernel(float4* __restrict__ out, int n4, uint32_t* __restrict__ cnt) {
    int i = blockIdx.x * blockDim.x + threadIdx.x;
    if (i < n4) out[i] = make_float4(0.f, 0.f, 0.f, 0.f);
    if (blockIdx.x == 0 && threadIdx.x < E_NUM) cnt[threadIdx.x] = 0u;
}

// ---------------------------------------------------------------------------
// Kernel 1: gating (softmax over E=8, top-2) + scatter tokens into expert
// buckets.  One wave32 per token.
// ---------------------------------------------------------------------------
__global__ void moe_gate_scatter_kernel(const float* __restrict__ x,
                                        const float* __restrict__ Wg,
                                        const float* __restrict__ bg,
                                        uint32_t* __restrict__ cnt,
                                        uint2* __restrict__ lists,
                                        int nTok) {
    int wave  = threadIdx.x >> 5;
    int lane  = threadIdx.x & 31;
    int token = blockIdx.x * (blockDim.x >> 5) + wave;
    if (token >= nTok) return;

    const float* xr = x + (size_t)token * D_DIM;
    float acc[E_NUM];
#pragma unroll
    for (int e = 0; e < E_NUM; ++e) acc[e] = 0.f;

    for (int i = 0; i < D_DIM / 32; ++i) {
        int d = i * 32 + lane;
        float xv = xr[d];
#pragma unroll
        for (int e = 0; e < E_NUM; ++e) acc[e] += xv * Wg[e * D_DIM + d];
    }
#pragma unroll
    for (int e = 0; e < E_NUM; ++e) {
#pragma unroll
        for (int off = 16; off >= 1; off >>= 1)
            acc[e] += __shfl_xor(acc[e], off, 32);
    }

    if (lane == 0) {
        float logits[E_NUM];
        float m = -3.0e38f;
#pragma unroll
        for (int e = 0; e < E_NUM; ++e) { logits[e] = acc[e] + bg[e]; m = fmaxf(m, logits[e]); }
        float sc[E_NUM];
        float ssum = 0.f;
#pragma unroll
        for (int e = 0; e < E_NUM; ++e) { sc[e] = __expf(logits[e] - m); ssum += sc[e]; }
        float inv = 1.f / ssum;
#pragma unroll
        for (int e = 0; e < E_NUM; ++e) sc[e] *= inv;

        int   i0 = 0; float s0 = sc[0];
#pragma unroll
        for (int e = 1; e < E_NUM; ++e) if (sc[e] > s0) { s0 = sc[e]; i0 = e; }
        int   i1 = -1; float s1 = -3.0e38f;
#pragma unroll
        for (int e = 0; e < E_NUM; ++e) if (e != i0 && sc[e] > s1) { s1 = sc[e]; i1 = e; }

        uint32_t p0 = atomicAdd(&cnt[i0], 1u);
        lists[(size_t)i0 * nTok + p0] = make_uint2((uint32_t)token, __float_as_uint(s0));
        uint32_t p1 = atomicAdd(&cnt[i1], 1u);
        lists[(size_t)i1 * nTok + p1] = make_uint2((uint32_t)token, __float_as_uint(s1));
    }
}

// ---------------------------------------------------------------------------
// Kernel 2: grouped GEMM over expert buckets, bf16 WMMA with fp32 accumulate.
// Block tile 128(M gathered tokens) x 128(N out cols), K-step 32, double-
// buffered LDS so next slab's global loads overlap WMMAs of the current slab.
// Epilogue: out[token,:] += score * (acc + bias[e,:])  via float atomics
// (exactly two commutative contributions per element -> deterministic).
// ---------------------------------------------------------------------------
__global__ __launch_bounds__(256) void moe_gemm_kernel(
    const float* __restrict__ x, const float* __restrict__ W,
    const float* __restrict__ b, const uint32_t* __restrict__ cnt,
    const uint2* __restrict__ lists, float* __restrict__ out,
    int nTok, int mtPerE) {

    __shared__ uint32_t ldsA[2][TILE_M * LDS_ROW_DW];
    __shared__ uint32_t ldsB[2][TILE_N * LDS_ROW_DW];
    __shared__ uint32_t s_tok[TILE_M];
    __shared__ float    s_scr[TILE_M];

    int e  = blockIdx.x / mtPerE;
    int mt = blockIdx.x % mtPerE;
    uint32_t rows = cnt[e];
    int m0 = mt * TILE_M;
    if ((uint32_t)m0 >= rows) return;          // empty tile -> early exit
    int n0  = blockIdx.y * TILE_N;
    int tid = threadIdx.x;

    if (tid < TILE_M) {
        int rg  = m0 + tid;
        int idx = (rg < (int)rows) ? rg : (int)rows - 1;   // clamp for padding rows
        uint2 ent = lists[(size_t)e * nTok + idx];
        s_tok[tid] = ent.x;
        s_scr[tid] = (rg < (int)rows) ? __uint_as_float(ent.y) : 0.f;
    }
    __syncthreads();

    int wv   = tid >> 5;
    int lane = tid & 31;
    int hl   = (lane >= 16) ? 1 : 0;
    int nlan = lane & 15;
    int mrow = wv * 16 + nlan;                 // LDS A row this lane's fragment reads

    // staging assignment: 2 threads per row, 16 fp32 (64B) each
    int srow   = tid >> 1;
    int shalf  = (tid & 1) * 16;               // fp32 offset within the 32-wide K slab
    int sdword = (tid & 1) * 8;                // dword offset in LDS row
    size_t   wRowBase = (size_t)e * O_DIM * D_DIM + (size_t)(n0 + srow) * D_DIM;
    uint32_t myTok    = s_tok[srow];
    const float* aSrcRow = x + (size_t)myTok * D_DIM + shalf;
    const float* wSrcRow = W + wRowBase + shalf;

    v8f acc[8] = {};
    union Frag { struct { uint4 lo; uint4 hi; } u; v16bf v; };

    // ---- prologue: stage k0 = 0 into buffer 0 ----
    {
        const float4* sa = (const float4*)(aSrcRow);
        const float4* sw = (const float4*)(wSrcRow);
        float4 a0 = sa[0], a1 = sa[1], a2 = sa[2], a3 = sa[3];
        float4 w0 = sw[0], w1 = sw[1], w2 = sw[2], w3 = sw[3];
        uint32_t* da = &ldsA[0][srow * LDS_ROW_DW + sdword];
        da[0] = pack_bf16_rne(a0.x, a0.y); da[1] = pack_bf16_rne(a0.z, a0.w);
        da[2] = pack_bf16_rne(a1.x, a1.y); da[3] = pack_bf16_rne(a1.z, a1.w);
        da[4] = pack_bf16_rne(a2.x, a2.y); da[5] = pack_bf16_rne(a2.z, a2.w);
        da[6] = pack_bf16_rne(a3.x, a3.y); da[7] = pack_bf16_rne(a3.z, a3.w);
        uint32_t* dw = &ldsB[0][srow * LDS_ROW_DW + sdword];
        dw[0] = pack_bf16_rne(w0.x, w0.y); dw[1] = pack_bf16_rne(w0.z, w0.w);
        dw[2] = pack_bf16_rne(w1.x, w1.y); dw[3] = pack_bf16_rne(w1.z, w1.w);
        dw[4] = pack_bf16_rne(w2.x, w2.y); dw[5] = pack_bf16_rne(w2.z, w2.w);
        dw[6] = pack_bf16_rne(w3.x, w3.y); dw[7] = pack_bf16_rne(w3.z, w3.w);
    }
    __syncthreads();

    for (int k0 = 0; k0 < D_DIM; k0 += TILE_K) {
        int  cur     = (k0 >> 5) & 1;
        int  nxt     = cur ^ 1;
        bool hasNext = (k0 + TILE_K) < D_DIM;

        // ---- issue global loads for the NEXT slab (overlap with WMMAs) ----
        float4 a0, a1, a2, a3, w0, w1, w2, w3;
        if (hasNext) {
            const float4* sa = (const float4*)(aSrcRow + k0 + TILE_K);
            const float4* sw = (const float4*)(wSrcRow + k0 + TILE_K);
            a0 = sa[0]; a1 = sa[1]; a2 = sa[2]; a3 = sa[3];
            w0 = sw[0]; w1 = sw[1]; w2 = sw[2]; w3 = sw[3];
        }

        // ---- A fragment: 16x32 bf16; lanes 0-15 K{0..7,16..23}, 16-31 K{8..15,24..31}
        Frag a;
        {
            const char* ar = (const char*)&ldsA[cur][mrow * LDS_ROW_DW] + hl * 16;
            a.u.lo = *(const uint4*)(ar);
            a.u.hi = *(const uint4*)(ar + 32);
        }
        // ---- all 8 B fragments first (pipelined dscnt waits), then 8 WMMAs ----
        Frag bfr[8];
#pragma unroll
        for (int t = 0; t < 8; ++t) {
            const char* br = (const char*)&ldsB[cur][(t * 16 + nlan) * LDS_ROW_DW] + hl * 16;
            bfr[t].u.lo = *(const uint4*)(br);
            bfr[t].u.hi = *(const uint4*)(br + 32);
        }
#pragma unroll
        for (int t = 0; t < 8; ++t) {
            acc[t] = __builtin_amdgcn_wmma_f32_16x16x32_bf16(
                false, a.v, false, bfr[t].v, (short)0, acc[t], false, false);
        }

        // ---- pack + store next slab into the other buffer ----
        if (hasNext) {
            uint32_t* da = &ldsA[nxt][srow * LDS_ROW_DW + sdword];
            da[0] = pack_bf16_rne(a0.x, a0.y); da[1] = pack_bf16_rne(a0.z, a0.w);
            da[2] = pack_bf16_rne(a1.x, a1.y); da[3] = pack_bf16_rne(a1.z, a1.w);
            da[4] = pack_bf16_rne(a2.x, a2.y); da[5] = pack_bf16_rne(a2.z, a2.w);
            da[6] = pack_bf16_rne(a3.x, a3.y); da[7] = pack_bf16_rne(a3.z, a3.w);
            uint32_t* dw = &ldsB[nxt][srow * LDS_ROW_DW + sdword];
            dw[0] = pack_bf16_rne(w0.x, w0.y); dw[1] = pack_bf16_rne(w0.z, w0.w);
            dw[2] = pack_bf16_rne(w1.x, w1.y); dw[3] = pack_bf16_rne(w1.z, w1.w);
            dw[4] = pack_bf16_rne(w2.x, w2.y); dw[5] = pack_bf16_rne(w2.z, w2.w);
            dw[6] = pack_bf16_rne(w3.x, w3.y); dw[7] = pack_bf16_rne(w3.z, w3.w);
            __syncthreads();
        }
    }

    // ---- epilogue: out[token, N] += score * (acc + bias) ----
    float bias[8];
#pragma unroll
    for (int t = 0; t < 8; ++t) bias[t] = b[e * O_DIM + n0 + t * 16 + nlan];

#pragma unroll
    for (int r = 0; r < 8; ++r) {
        int Ml = wv * 16 + r + hl * 8;         // C layout: lanes>=16 hold M = r+8
        int rg = m0 + Ml;
        if (rg < (int)rows) {
            uint32_t token = s_tok[Ml];
            float    s     = s_scr[Ml];
            float* orow = out + (size_t)token * O_DIM;
#pragma unroll
            for (int t = 0; t < 8; ++t) {
                int N = n0 + t * 16 + nlan;
                float v = s * (acc[t][r] + bias[t]);
                atomicAdd(&orow[N], v);
            }
        }
    }
}

// ---------------------------------------------------------------------------
extern "C" void kernel_launch(void* const* d_in, const int* in_sizes, int n_in,
                              void* d_out, int out_size, void* d_ws, size_t ws_size,
                              hipStream_t stream) {
    (void)n_in; (void)ws_size;
    const float* x  = (const float*)d_in[0];   // [B,S,D]
    const float* W  = (const float*)d_in[1];   // [E,O,D]
    const float* b  = (const float*)d_in[2];   // [E,O]
    const float* Wg = (const float*)d_in[3];   // [E,D]
    const float* bg = (const float*)d_in[4];   // [E]
    float* out = (float*)d_out;                // [B,S,O]

    int nTok = in_sizes[0] / D_DIM;            // 8192

    uint32_t* cnt   = (uint32_t*)d_ws;                 // 8 counters
    uint2*    lists = (uint2*)((char*)d_ws + 256);     // E * nTok entries (token, score)

    int n4 = out_size / 4;
    moe_zero_kernel<<<(n4 + 255) / 256, 256, 0, stream>>>((float4*)out, n4, cnt);

    moe_gate_scatter_kernel<<<(nTok + 7) / 8, 256, 0, stream>>>(x, Wg, bg, cnt, lists, nTok);

    int mtPerE = (nTok + TILE_M - 1) / TILE_M;         // worst case all tokens -> one expert
    dim3 grid(E_NUM * mtPerE, O_DIM / TILE_N);
    moe_gemm_kernel<<<grid, 256, 0, stream>>>(x, W, b, cnt, lists, out, nTok, mtPerE);
}
